// GAT_46505905881808
// MI455X (gfx1250) — compile-verified, matrix operand order checked by reading, not run on hardware
//
#include <hip/hip_runtime.h>
#include <hip/hip_bf16.h>

// ---------------------------------------------------------------------------
// GAT 3-layer forward for MI455X (gfx1250, wave32, WMMA).
// GEMMs: bf16 WMMA (v_wmma_f32_16x16x32_bf16), f32 accumulate.
// Edge phase: segment softmax + scatter with f32 atomics (memory bound).
// ---------------------------------------------------------------------------

#define NNODES 50000
#define NEDGES 1600000
#define IN_CH  128
#define HID    32
#define HEADS  4
#define OUT_CH 64
#define NEG_SLOPE 0.2f

typedef __bf16 bf16_t;
typedef __attribute__((ext_vector_type(16))) __bf16       v16bf;
typedef __attribute__((ext_vector_type(8)))  float        v8f;
typedef __attribute__((ext_vector_type(4)))  unsigned int v4u;

static inline int cdiv(long long a, long long b) { return (int)((a + b - 1) / b); }

// ------------------------- conversion kernels ------------------------------

__global__ __launch_bounds__(256) void k_f32_to_bf16(
    const float* __restrict__ in, bf16_t* __restrict__ out, long long n) {
  long long gid = (long long)blockIdx.x * blockDim.x + threadIdx.x;
  if (gid < n) out[gid] = (bf16_t)in[gid];
}

// Wt[n][k] = (bf16) W[k][n]   (W is [K, NC] row-major)
__global__ __launch_bounds__(256) void k_transpose_bf16(
    const float* __restrict__ W, bf16_t* __restrict__ Wt, int K, int NC) {
  int gid = blockIdx.x * blockDim.x + threadIdx.x;
  if (gid >= K * NC) return;
  int k = gid / NC;
  int n = gid - k * NC;
  Wt[(long long)n * K + k] = (bf16_t)W[gid];
}

// ------------------------- WMMA bf16 GEMM ----------------------------------
// C[M,NC] = A[M,K] * B[K,NC], B given transposed as Bt[NC,K]. One wave per
// 16x16 C tile. A/B fragment lane layout per CDNA5 ISA 16-bit A 16x32:
// lanes 0-15: K = 0..7, 16..23 ; lanes 16-31: K = 8..15, 24..31 (base kb).
__global__ __launch_bounds__(256) void k_gemm_bf16_wmma(
    const bf16_t* __restrict__ A, const bf16_t* __restrict__ Bt,
    float* __restrict__ C, int M, int K, int NC) {
  const int wave   = (blockIdx.x * blockDim.x + threadIdx.x) >> 5;
  const int lane   = threadIdx.x & 31;
  const int ntiles = NC >> 4;
  const int total  = (M >> 4) * ntiles;
  if (wave >= total) return;
  const int mt   = wave / ntiles;
  const int nt   = wave - mt * ntiles;
  const int half = lane >> 4;   // 0: K-low pattern, 1: K-high pattern
  const int l16  = lane & 15;
  const int off  = half * 8;

  const bf16_t* ap = A  + (long long)(mt * 16 + l16) * K + off;
  const bf16_t* bp = Bt + (long long)(nt * 16 + l16) * K + off;

  v8f acc = {};
#pragma unroll
  for (int kb = 0; kb < 128; kb += 32) {
    union { v16bf v; v4u u[2]; } fa, fb;
    fa.u[0] = *(const v4u*)(ap + kb);
    fa.u[1] = *(const v4u*)(ap + kb + 16);
    fb.u[0] = *(const v4u*)(bp + kb);
    fb.u[1] = *(const v4u*)(bp + kb + 16);
    acc = __builtin_amdgcn_wmma_f32_16x16x32_bf16(
        false, fa.v, false, fb.v, (short)0, acc, false, false);
  }
  // C/D layout: VGPR r -> row (mt*16 + r + 8*half), col nt*16 + l16
  float* cp = C + (long long)(mt * 16 + half * 8) * NC + nt * 16 + l16;
#pragma unroll
  for (int r = 0; r < 8; ++r) cp[(long long)r * NC] = acc[r];
}

// ------------------------- attention scores --------------------------------
// as_[n,h] = dot(h[n,h,:], a_src[h,:]) ; ad_ likewise. One thread per (n,h).
__global__ __launch_bounds__(256) void k_node_scores(
    const float* __restrict__ h, const float* __restrict__ a_src,
    const float* __restrict__ a_dst, float* __restrict__ as_,
    float* __restrict__ ad_, int C, int n_nodes) {
  int gid = blockIdx.x * blockDim.x + threadIdx.x;
  if (gid >= n_nodes * 4) return;
  int n  = gid >> 2;
  int hd = gid & 3;
  int TOT = 4 * C;
  const float* hp = h + (long long)n * TOT + hd * C;
  const float* sp = a_src + hd * C;
  const float* dp = a_dst + hd * C;
  float r1 = 0.f, r2 = 0.f;
  for (int c = 0; c < C; c += 4) {
    float4 hv = *(const float4*)(hp + c);
    float4 sv = *(const float4*)(sp + c);
    float4 dv = *(const float4*)(dp + c);
    r1 += hv.x * sv.x + hv.y * sv.y + hv.z * sv.z + hv.w * sv.w;
    r2 += hv.x * dv.x + hv.y * dv.y + hv.z * dv.z + hv.w * dv.w;
  }
  as_[gid] = r1;
  ad_[gid] = r2;
}

// ------------------------- segment softmax ---------------------------------
__device__ __forceinline__ unsigned fkey(float x) {
  unsigned u = __float_as_uint(x);
  return (u & 0x80000000u) ? ~u : (u | 0x80000000u);
}
__device__ __forceinline__ float funkey(unsigned k) {
  unsigned u = (k & 0x80000000u) ? (k & 0x7fffffffu) : ~k;
  return __uint_as_float(u);
}
__device__ __forceinline__ float lrelu(float x) {
  return x > 0.f ? x : NEG_SLOPE * x;
}

// pass 1: e = leaky_relu(as[src] + ad[dst]); alpha <- e; segment max (ordered keys)
__global__ __launch_bounds__(256) void k_edge_score_max(
    const long long* __restrict__ ei, const float* __restrict__ as_,
    const float* __restrict__ ad_, float* __restrict__ ebuf,
    unsigned* __restrict__ mkey, int nE) {
  int e = blockIdx.x * blockDim.x + threadIdx.x;
  if (e >= nE) return;
  int s = (int)ei[e];
  int d = (int)ei[(long long)nE + e];
  float4 a = ((const float4*)as_)[s];
  float4 b = ((const float4*)ad_)[d];
  float4 sc;
  sc.x = lrelu(a.x + b.x);
  sc.y = lrelu(a.y + b.y);
  sc.z = lrelu(a.z + b.z);
  sc.w = lrelu(a.w + b.w);
  ((float4*)ebuf)[e] = sc;
  unsigned* mp = mkey + (long long)d * 4;
  atomicMax(mp + 0, fkey(sc.x));
  atomicMax(mp + 1, fkey(sc.y));
  atomicMax(mp + 2, fkey(sc.z));
  atomicMax(mp + 3, fkey(sc.w));
}

// pass 2: ex = exp(e - m[dst]); alpha <- ex; segment sum
__global__ __launch_bounds__(256) void k_edge_exp_sum(
    const long long* __restrict__ ei, float* __restrict__ ebuf,
    const unsigned* __restrict__ mkey, float* __restrict__ ssum, int nE) {
  int e = blockIdx.x * blockDim.x + threadIdx.x;
  if (e >= nE) return;
  int d = (int)ei[(long long)nE + e];
  float4 sc = ((const float4*)ebuf)[e];
  const unsigned* mp = mkey + (long long)d * 4;
  float4 ex;
  ex.x = __expf(sc.x - funkey(mp[0]));
  ex.y = __expf(sc.y - funkey(mp[1]));
  ex.z = __expf(sc.z - funkey(mp[2]));
  ex.w = __expf(sc.w - funkey(mp[3]));
  ((float4*)ebuf)[e] = ex;
  float* sp = ssum + (long long)d * 4;
  atomicAdd(sp + 0, ex.x);
  atomicAdd(sp + 1, ex.y);
  atomicAdd(sp + 2, ex.z);
  atomicAdd(sp + 3, ex.w);
}

// pass 3: alpha = ex / (s[dst]+eps); write alpha; agg[dst] += h[src] * alpha.
// One wave per edge; lane covers 4 channels per 128-channel stripe.
__global__ __launch_bounds__(256) void k_edge_norm_scatter(
    const long long* __restrict__ ei, const float* __restrict__ h,
    const float* __restrict__ ssum, float* __restrict__ alpha,
    float* __restrict__ agg, int C, int nE) {
  int wid  = (blockIdx.x * blockDim.x + threadIdx.x) >> 5;
  int lane = threadIdx.x & 31;
  if (wid >= nE) return;
  int s = (int)ei[wid];
  int d = (int)ei[(long long)nE + wid];
  int TOT = 4 * C;
  const float* hp = h + (long long)s * TOT;
  float* ap = agg + (long long)d * TOT;
  for (int c0 = lane * 4; c0 < TOT; c0 += 128) {
    int hd = c0 / C;
    float ex = alpha[(long long)wid * 4 + hd];
    float a  = ex / (ssum[(long long)d * 4 + hd] + 1e-16f);
    if ((c0 & (C - 1)) == 0) alpha[(long long)wid * 4 + hd] = a;  // once per head
    float4 hv = *(const float4*)(hp + c0);
    atomicAdd(ap + c0 + 0, hv.x * a);
    atomicAdd(ap + c0 + 1, hv.y * a);
    atomicAdd(ap + c0 + 2, hv.z * a);
    atomicAdd(ap + c0 + 3, hv.w * a);
  }
}

// ------------------------- node post: ELU + LayerNorm (+skip) --------------
// One wave per node, 128 channels, lane handles 4 via float4.
__global__ __launch_bounds__(256) void k_node_post_ln(
    const float* __restrict__ agg, const float* __restrict__ bias,
    const float* __restrict__ lnw, const float* __restrict__ lnb,
    const float* __restrict__ skip, const float* __restrict__ bskip,
    float* __restrict__ out, int n_nodes, int has_skip) {
  int wid  = (blockIdx.x * blockDim.x + threadIdx.x) >> 5;
  int lane = threadIdx.x & 31;
  if (wid >= n_nodes) return;
  float4 v  = ((const float4*)(agg + (long long)wid * 128))[lane];
  float4 bb = ((const float4*)bias)[lane];
  v.x += bb.x; v.y += bb.y; v.z += bb.z; v.w += bb.w;
  v.x = v.x > 0.f ? v.x : expm1f(v.x);
  v.y = v.y > 0.f ? v.y : expm1f(v.y);
  v.z = v.z > 0.f ? v.z : expm1f(v.z);
  v.w = v.w > 0.f ? v.w : expm1f(v.w);
  float sm = v.x + v.y + v.z + v.w;
#pragma unroll
  for (int o = 16; o; o >>= 1) sm += __shfl_xor(sm, o, 32);
  float mu = sm * (1.f / 128.f);
  float4 dl;
  dl.x = v.x - mu; dl.y = v.y - mu; dl.z = v.z - mu; dl.w = v.w - mu;
  float q = dl.x * dl.x + dl.y * dl.y + dl.z * dl.z + dl.w * dl.w;
#pragma unroll
  for (int o = 16; o; o >>= 1) q += __shfl_xor(q, o, 32);
  float r = rsqrtf(q * (1.f / 128.f) + 1e-5f);
  float4 w4 = ((const float4*)lnw)[lane];
  float4 lb = ((const float4*)lnb)[lane];
  float4 o4;
  o4.x = dl.x * r * w4.x + lb.x;
  o4.y = dl.y * r * w4.y + lb.y;
  o4.z = dl.z * r * w4.z + lb.z;
  o4.w = dl.w * r * w4.w + lb.w;
  if (has_skip) {
    float4 sk = ((const float4*)(skip + (long long)wid * 128))[lane];
    float4 bs = ((const float4*)bskip)[lane];
    o4.x += sk.x + bs.x; o4.y += sk.y + bs.y;
    o4.z += sk.z + bs.z; o4.w += sk.w + bs.w;
  }
  ((float4*)(out + (long long)wid * 128))[lane] = o4;
}

// layer-3 finalize: out[n,c] = mean_h agg[n,h,c] + b3[c]
__global__ __launch_bounds__(256) void k_finalize_mean(
    const float* __restrict__ agg, const float* __restrict__ b3,
    float* __restrict__ out, int n_nodes) {
  int gid = blockIdx.x * blockDim.x + threadIdx.x;
  if (gid >= n_nodes * 64) return;
  int n = gid >> 6;
  int c = gid & 63;
  const float* a = agg + (long long)n * 256 + c;
  out[gid] = 0.25f * (a[0] + a[64] + a[128] + a[192]) + b3[c];
}

// ---------------------------------------------------------------------------

extern "C" void kernel_launch(void* const* d_in, const int* in_sizes, int n_in,
                              void* d_out, int out_size, void* d_ws, size_t ws_size,
                              hipStream_t stream) {
  (void)in_sizes; (void)n_in; (void)out_size; (void)ws_size;
  const float*     x      = (const float*)d_in[0];
  const long long* ei     = (const long long*)d_in[1];   // int64 [2,E]
  const float*     W1     = (const float*)d_in[2];
  const float*     a_src1 = (const float*)d_in[3];
  const float*     a_dst1 = (const float*)d_in[4];
  const float*     b1     = (const float*)d_in[5];
  const float*     ln1w   = (const float*)d_in[6];
  const float*     ln1b   = (const float*)d_in[7];
  const float*     Wskip  = (const float*)d_in[8];
  const float*     bskip  = (const float*)d_in[9];
  const float*     W2     = (const float*)d_in[10];
  const float*     a_src2 = (const float*)d_in[11];
  const float*     a_dst2 = (const float*)d_in[12];
  const float*     b2     = (const float*)d_in[13];
  const float*     ln2w   = (const float*)d_in[14];
  const float*     ln2b   = (const float*)d_in[15];
  const float*     W3     = (const float*)d_in[16];
  const float*     a_src3 = (const float*)d_in[17];
  const float*     a_dst3 = (const float*)d_in[18];
  const float*     b3     = (const float*)d_in[19];

  float* out    = (float*)d_out;                       // [N,64]
  float* alpha1 = out + (long long)NNODES * 64;        // [E,4]
  float* alpha2 = alpha1 + (long long)NEDGES * 4;
  float* alpha3 = alpha2 + (long long)NEDGES * 4;

  // ---- workspace bump allocator (256B aligned) ----
  char* p = (char*)d_ws;
  auto alloc = [&](size_t bytes) -> char* {
    char* r = p;
    p += (bytes + 255) & ~(size_t)255;
    return r;
  };
  bf16_t*   xb    = (bf16_t*)alloc((size_t)NNODES * 128 * 2);
  bf16_t*   hb    = (bf16_t*)alloc((size_t)NNODES * 128 * 2);
  bf16_t*   W1t   = (bf16_t*)alloc(128 * 128 * 2);
  bf16_t*   Wst   = (bf16_t*)alloc(128 * 128 * 2);
  bf16_t*   W2t   = (bf16_t*)alloc(128 * 128 * 2);
  bf16_t*   W3t   = (bf16_t*)alloc(128 * 256 * 2);
  float*    hraw  = (float*)alloc((size_t)NNODES * 256 * 4);   // layer3 needs 256
  float*    skip  = (float*)alloc((size_t)NNODES * 128 * 4);
  float*    hpost = (float*)alloc((size_t)NNODES * 128 * 4);
  float*    agg   = (float*)alloc((size_t)NNODES * 256 * 4);
  float*    as_   = (float*)alloc((size_t)NNODES * 4 * 4);
  float*    ad_   = (float*)alloc((size_t)NNODES * 4 * 4);
  unsigned* mkey  = (unsigned*)alloc((size_t)NNODES * 4 * 4);
  float*    ssum  = (float*)alloc((size_t)NNODES * 4 * 4);

  const int B = 256;
  const int gemm_blocks128 = cdiv((long long)(NNODES / 16) * 8 * 32, B);   // 128 cols
  const int gemm_blocks256 = cdiv((long long)(NNODES / 16) * 16 * 32, B);  // 256 cols
  const int edge_blocks    = cdiv(NEDGES, B);
  const int edge_wave_blks = cdiv((long long)NEDGES * 32, B);
  const int node_wave_blks = cdiv((long long)NNODES * 32, B);

  // ---- weight + input conversion ----
  k_f32_to_bf16<<<cdiv((long long)NNODES * 128, B), B, 0, stream>>>(x, xb, (long long)NNODES * 128);
  k_transpose_bf16<<<cdiv(128 * 128, B), B, 0, stream>>>(W1, W1t, 128, 128);
  k_transpose_bf16<<<cdiv(128 * 128, B), B, 0, stream>>>(Wskip, Wst, 128, 128);
  k_transpose_bf16<<<cdiv(128 * 128, B), B, 0, stream>>>(W2, W2t, 128, 128);
  k_transpose_bf16<<<cdiv(128 * 256, B), B, 0, stream>>>(W3, W3t, 128, 256);

  // =========================== layer 1 ===========================
  k_gemm_bf16_wmma<<<gemm_blocks128, B, 0, stream>>>(xb, W1t, hraw, NNODES, 128, 128);
  k_gemm_bf16_wmma<<<gemm_blocks128, B, 0, stream>>>(xb, Wst, skip, NNODES, 128, 128);
  k_node_scores<<<cdiv(NNODES * 4, B), B, 0, stream>>>(hraw, a_src1, a_dst1, as_, ad_, HID, NNODES);
  hipMemsetAsync(mkey, 0, (size_t)NNODES * 16, stream);
  hipMemsetAsync(ssum, 0, (size_t)NNODES * 16, stream);
  hipMemsetAsync(agg, 0, (size_t)NNODES * 128 * 4, stream);
  k_edge_score_max<<<edge_blocks, B, 0, stream>>>(ei, as_, ad_, alpha1, mkey, NEDGES);
  k_edge_exp_sum<<<edge_blocks, B, 0, stream>>>(ei, alpha1, mkey, ssum, NEDGES);
  k_edge_norm_scatter<<<edge_wave_blks, B, 0, stream>>>(ei, hraw, ssum, alpha1, agg, HID, NEDGES);
  k_node_post_ln<<<node_wave_blks, B, 0, stream>>>(agg, b1, ln1w, ln1b, skip, bskip, hpost, NNODES, 1);
  k_f32_to_bf16<<<cdiv((long long)NNODES * 128, B), B, 0, stream>>>(hpost, hb, (long long)NNODES * 128);

  // =========================== layer 2 ===========================
  k_gemm_bf16_wmma<<<gemm_blocks128, B, 0, stream>>>(hb, W2t, hraw, NNODES, 128, 128);
  k_node_scores<<<cdiv(NNODES * 4, B), B, 0, stream>>>(hraw, a_src2, a_dst2, as_, ad_, HID, NNODES);
  hipMemsetAsync(mkey, 0, (size_t)NNODES * 16, stream);
  hipMemsetAsync(ssum, 0, (size_t)NNODES * 16, stream);
  hipMemsetAsync(agg, 0, (size_t)NNODES * 128 * 4, stream);
  k_edge_score_max<<<edge_blocks, B, 0, stream>>>(ei, as_, ad_, alpha2, mkey, NEDGES);
  k_edge_exp_sum<<<edge_blocks, B, 0, stream>>>(ei, alpha2, mkey, ssum, NEDGES);
  k_edge_norm_scatter<<<edge_wave_blks, B, 0, stream>>>(ei, hraw, ssum, alpha2, agg, HID, NEDGES);
  k_node_post_ln<<<node_wave_blks, B, 0, stream>>>(agg, b2, ln2w, ln2b, nullptr, nullptr, hpost, NNODES, 0);
  k_f32_to_bf16<<<cdiv((long long)NNODES * 128, B), B, 0, stream>>>(hpost, hb, (long long)NNODES * 128);

  // =========================== layer 3 ===========================
  k_gemm_bf16_wmma<<<gemm_blocks256, B, 0, stream>>>(hb, W3t, hraw, NNODES, 128, 256);
  k_node_scores<<<cdiv(NNODES * 4, B), B, 0, stream>>>(hraw, a_src3, a_dst3, as_, ad_, OUT_CH, NNODES);
  hipMemsetAsync(mkey, 0, (size_t)NNODES * 16, stream);
  hipMemsetAsync(ssum, 0, (size_t)NNODES * 16, stream);
  hipMemsetAsync(agg, 0, (size_t)NNODES * 256 * 4, stream);
  k_edge_score_max<<<edge_blocks, B, 0, stream>>>(ei, as_, ad_, alpha3, mkey, NEDGES);
  k_edge_exp_sum<<<edge_blocks, B, 0, stream>>>(ei, alpha3, mkey, ssum, NEDGES);
  k_edge_norm_scatter<<<edge_wave_blks, B, 0, stream>>>(ei, hraw, ssum, alpha3, agg, OUT_CH, NEDGES);
  k_finalize_mean<<<cdiv(NNODES * 64, B), B, 0, stream>>>(agg, b3, out, NNODES);
}
